// DecoderCellContent_86028194939187
// MI455X (gfx1250) — matrix operand order repaired and checked by reading, not compiled, not run on hardware
//
#include <hip/hip_runtime.h>

typedef _Float16 half_t;
typedef __attribute__((ext_vector_type(16))) _Float16 v16h;
typedef __attribute__((ext_vector_type(8)))  _Float16 v8h;
typedef __attribute__((ext_vector_type(8)))  float    v8f;

#define B_   128
#define N_   400
#define E_   512
#define SH_  256
#define H_   512
#define A_   256
#define EMB_ 256
#define V_   1000
#define T_   256

#define XDIM_   1280   // x_cat row: [context(512) | emb(256) | h(512)]
#define XH_OFF_ 768    // offset of h inside x_cat row
#define NG_     2048   // combined gate GEMM output columns

// ---------------------------------------------------------------------------
// Preamble helpers
// ---------------------------------------------------------------------------
__global__ void f32_to_f16_kernel(const float* __restrict__ in,
                                  half_t* __restrict__ out, long n) {
  long i = (long)blockIdx.x * blockDim.x + threadIdx.x;
  long stride = (long)gridDim.x * blockDim.x;
  for (; i < n; i += stride) out[i] = (half_t)in[i];
}

// out[c*R + r] = in[r*C + c]   (LDS-tiled transpose, f32 -> f16)
__global__ __launch_bounds__(256)
void transpose_f32_to_f16_kernel(const float* __restrict__ in,
                                 half_t* __restrict__ out, int R, int C) {
  __shared__ float tile[32][33];
  int tx = threadIdx.x & 31, ty = threadIdx.x >> 5;
  int r0 = blockIdx.y * 32, c0 = blockIdx.x * 32;
  for (int i = ty; i < 32; i += 8) {
    int r = r0 + i, c = c0 + tx;
    tile[i][tx] = (r < R && c < C) ? in[(long)r * C + c] : 0.f;
  }
  __syncthreads();
  for (int i = ty; i < 32; i += 8) {
    int c = c0 + i, r = r0 + tx;
    if (c < C && r < R) out[(long)c * R + r] = (half_t)tile[tx][i];
  }
}

// Combined gate weight: Wbig[n,k], n<1024: [W_ih[n] | W_hh[n]] (r,z gates);
// 1024..1535: [W_ih[n] | 0] (gi_n); 1536..2047: [0 | W_hh[n-512]] (gh_n).
__global__ void build_wbig_kernel(const float* __restrict__ W_ih,
                                  const float* __restrict__ W_hh,
                                  const float* __restrict__ b_ih,
                                  const float* __restrict__ b_hh,
                                  half_t* __restrict__ Wbig,
                                  float* __restrict__ bbig) {
  long idx = (long)blockIdx.x * blockDim.x + threadIdx.x;
  long stride = (long)gridDim.x * blockDim.x;
  const long total = (long)NG_ * XDIM_;
  for (; idx < total; idx += stride) {
    int n = (int)(idx / XDIM_), k = (int)(idx % XDIM_);
    float v;
    if (n < 1024)      v = (k < XH_OFF_) ? W_ih[(long)n * XH_OFF_ + k]
                                         : W_hh[(long)n * H_ + (k - XH_OFF_)];
    else if (n < 1536) v = (k < XH_OFF_) ? W_ih[(long)n * XH_OFF_ + k] : 0.f;
    else               v = (k >= XH_OFF_) ? W_hh[(long)(n - 512) * H_ + (k - XH_OFF_)] : 0.f;
    Wbig[idx] = (half_t)v;
    if (k == 0)
      bbig[n] = (n < 1024) ? (b_ih[n] + b_hh[n])
                           : ((n < 1536) ? b_ih[n] : b_hh[n - 512]);
  }
}

// zero h region of x_cat + loss accumulator (every call: determinism)
__global__ void init_state_kernel(half_t* __restrict__ x_cat, float* __restrict__ acc) {
  int i = blockIdx.x * blockDim.x + threadIdx.x;
  if (i < B_ * H_) {
    int b = i / H_, j = i % H_;
    x_cat[(long)b * XDIM_ + XH_OFF_ + j] = (half_t)0.f;
  }
  if (i == 0) acc[0] = 0.f;
}

// ---------------------------------------------------------------------------
// WMMA GEMM:  C[M,N] = A[M,K](f16 rm) @ Bw[N,K]^T(f16 rm) + bias
// Block = 4 waves = 64x64 output tile. B tiles (64 cols x 64 K, f16) staged
// in LDS (double-buffered, stride 72 halfs -> conflict-free b128 banks).
// Per K-tile: A frags issued first (their wait leaves the next-tile global
// loads in flight); the 8 LDS fragment reads are ping-pong pipelined two
// deep so each WMMA's ds-wait leaves the next fragment's loads outstanding.
// One barrier per tile. K must be a multiple of 64.
// ---------------------------------------------------------------------------
#define BK_   64
#define LDSB_ 72   // padded LDS row stride (halfs)

__global__ __launch_bounds__(128)
void wmma_gemm_kernel(const half_t* __restrict__ A, const half_t* __restrict__ Bw,
                      const float* __restrict__ bias,
                      float* __restrict__ outF, half_t* __restrict__ outH,
                      int M, int N, int K, int lda, int ldb, int ldc) {
  __shared__ half_t Bs[2][64 * LDSB_];

  const int tid  = threadIdx.x;
  const int wave = tid >> 5, lane = tid & 31;
  const int m0 = (blockIdx.y * 4 + wave) * 16;
  const int n0 = blockIdx.x * 64;

  // cooperative B-tile loader: thread -> (col 0..63, K-half 0/32)
  const int lcol = tid >> 1;
  const int lk   = (tid & 1) * 32;
  int gcol = n0 + lcol; if (gcol > N - 1) gcol = N - 1;
  const half_t* gB = Bw + (long)gcol * ldb + lk;
  const int soff = lcol * LDSB_ + lk;

  // per-lane A fragment / LDS B fragment offsets (16-bit WMMA layouts)
  const int  mrow = m0 + (lane & 15);
  const half_t* Arow = A + (long)mrow * lda;
  const int aoff = (lane >> 4) * 16;
  const int koff = (lane >> 4) * 16;

  v8f acc[4] = {};
  union HV { v16h v; v8h h[2]; };

  // preload K-tile 0 (64 K wide): 2 x 32B per thread
  HV t0, t1;
  t0.h[0] = *(const v8h*)(gB);      t0.h[1] = *(const v8h*)(gB + 8);
  t1.h[0] = *(const v8h*)(gB + 16); t1.h[1] = *(const v8h*)(gB + 24);
  {
    half_t* sB = &Bs[0][soff];
    *(v8h*)(sB)      = t0.h[0];
    *(v8h*)(sB + 8)  = t0.h[1];
    *(v8h*)(sB + 16) = t1.h[0];
    *(v8h*)(sB + 24) = t1.h[1];
  }
  __syncthreads();

  int buf = 0;
  for (int kb = 0; kb < K; kb += BK_) {
    const bool has_next = (kb + BK_) < K;
    // A fragments for both 32-K sub-steps -- issued BEFORE the next-tile
    // global loads so waiting on A leaves the tile prefetch outstanding.
    HV a0, a1;
    a0.h[0] = *(const v8h*)(Arow + kb + aoff);
    a0.h[1] = *(const v8h*)(Arow + kb + 8 + aoff);
    a1.h[0] = *(const v8h*)(Arow + kb + 32 + aoff);
    a1.h[1] = *(const v8h*)(Arow + kb + 40 + aoff);
    if (has_next) {
      const half_t* gn = gB + kb + BK_;
      t0.h[0] = *(const v8h*)(gn);      t0.h[1] = *(const v8h*)(gn + 8);
      t1.h[0] = *(const v8h*)(gn + 16); t1.h[1] = *(const v8h*)(gn + 24);
    }

    // 8 WMMAs (2 sub-steps x 4 col tiles) with 2-deep LDS fragment pipeline.
    auto ldB = [&](int u) -> HV {
      const int kk = u >> 2, s = u & 3;
      const half_t* bp =
          &Bs[buf][(s * 16 + (lane & 15)) * LDSB_ + kk * 32 + koff];
      HV r;
      r.h[0] = *(const v8h*)(bp);
      r.h[1] = *(const v8h*)(bp + 8);
      return r;
    };
    HV bcur = ldB(0), bnext;
#pragma unroll
    for (int u = 0; u < 8; ++u) {
      if (u < 7) bnext = ldB(u + 1);
      const HV& a = (u >> 2) ? a1 : a0;
      const int s = u & 3;
      acc[s] = __builtin_amdgcn_wmma_f32_16x16x32_f16(
          false, a.v, false, bcur.v, (short)0, acc[s], false, false);
      bcur = bnext;
    }

    if (has_next) {
      half_t* sB = &Bs[buf ^ 1][soff];
      *(v8h*)(sB)      = t0.h[0];
      *(v8h*)(sB + 8)  = t0.h[1];
      *(v8h*)(sB + 16) = t1.h[0];
      *(v8h*)(sB + 24) = t1.h[1];
    }
    __syncthreads();
    buf ^= 1;
  }

#pragma unroll
  for (int s = 0; s < 4; ++s) {
    int col = n0 + s * 16 + (lane & 15);
    if (col >= N) continue;
    float bv = bias ? bias[col] : 0.f;
#pragma unroll
    for (int v = 0; v < 8; ++v) {
      int row = m0 + v + (lane >> 4) * 8;   // C/D layout: VGPR v -> M=v / M=8+v
      if (row >= M) continue;
      float val = acc[s][v] + bv;
      if (outF) outF[(long)row * ldc + col] = val;
      if (outH) outH[(long)row * ldc + col] = (half_t)val;
    }
  }
}

// ---------------------------------------------------------------------------
// Fused attention step (one block per batch element, 256 threads):
//   s = sproj + bc + h@Wc   (h staged in LDS, per-thread dot vs Wc_t row)
//   scores = tanh(feat_proj + s) . v_att ; softmax over N
//   context = alpha @ features ; write [context | emb(prev tok)] into x_cat
// ---------------------------------------------------------------------------
__global__ __launch_bounds__(256)
void attention_kernel(const half_t* __restrict__ feat_proj,  // [B*N, A]
                      const half_t* __restrict__ feats,      // [B*N, E]
                      const float*  __restrict__ sproj,      // [B, A] (incl bs)
                      const float*  __restrict__ bc,         // [A]
                      const half_t* __restrict__ Wc_t,       // [A, H]
                      const float*  __restrict__ v_att,      // [A]
                      const int*    __restrict__ tgt,        // [B, T]
                      const half_t* __restrict__ emb,        // [V, EMB]
                      half_t* __restrict__ x_cat,            // [B, XDIM]
                      int t) {
  __shared__ float s_h[H_], s_s[A_], s_v[A_], s_e[N_], red[256];
  const int b = blockIdx.x, tid = threadIdx.x;

  const half_t* hrow = x_cat + (long)b * XDIM_ + XH_OFF_;
  s_h[tid]       = (float)hrow[tid];
  s_h[tid + 256] = (float)hrow[tid + 256];
  s_v[tid] = v_att[tid];
  __syncthreads();

  {  // s[a] = sproj + bc + h . Wc_t[a,:]
    const half_t* wrow = Wc_t + (long)tid * H_;
    float accv = 0.f;
    for (int k = 0; k < H_; k += 8) {
      v8h w8 = *(const v8h*)(wrow + k);
#pragma unroll
      for (int u = 0; u < 8; ++u) accv += s_h[k + u] * (float)w8[u];
    }
    s_s[tid] = sproj[b * A_ + tid] + bc[tid] + accv;
  }
  __syncthreads();

  for (int n = tid; n < N_; n += 256) {
    const half_t* fp = feat_proj + ((long)b * N_ + n) * A_;
    float e = 0.f;
    for (int a = 0; a < A_; a += 8) {
      v8h f8 = *(const v8h*)(fp + a);
#pragma unroll
      for (int u = 0; u < 8; ++u)
        e += tanhf((float)f8[u] + s_s[a + u]) * s_v[a + u];
    }
    s_e[n] = e;
  }
  __syncthreads();

  float m = -3.4e38f;
  for (int n = tid; n < N_; n += 256) m = fmaxf(m, s_e[n]);
  red[tid] = m; __syncthreads();
  for (int o = 128; o > 0; o >>= 1) {
    if (tid < o) red[tid] = fmaxf(red[tid], red[tid + o]);
    __syncthreads();
  }
  m = red[0]; __syncthreads();

  float sum = 0.f;
  for (int n = tid; n < N_; n += 256) {
    float ev = expf(s_e[n] - m);
    s_e[n] = ev;
    sum += ev;
  }
  red[tid] = sum; __syncthreads();
  for (int o = 128; o > 0; o >>= 1) {
    if (tid < o) red[tid] += red[tid + o];
    __syncthreads();
  }
  float inv = 1.f / red[0];
  __syncthreads();
  for (int n = tid; n < N_; n += 256) s_e[n] *= inv;
  __syncthreads();

  for (int e0 = tid; e0 < E_; e0 += 256) {
    float c = 0.f;
    const half_t* fb = feats + (long)b * N_ * E_ + e0;
    for (int n = 0; n < N_; ++n) c += s_e[n] * (float)fb[(long)n * E_];
    x_cat[(long)b * XDIM_ + e0] = (half_t)c;
  }
  int tok = (t == 0) ? 0 : tgt[b * T_ + (t - 1)];   // teacher forcing, START=0
  x_cat[(long)b * XDIM_ + E_ + tid] = emb[(long)tok * EMB_ + tid];
}

// ---------------------------------------------------------------------------
// GRU gates from combined GEMM output G[B, 2048]; h updated in-place in x_cat
// ---------------------------------------------------------------------------
__global__ __launch_bounds__(256)
void gru_gate_kernel(const float* __restrict__ G, half_t* __restrict__ x_cat) {
  int idx = blockIdx.x * 256 + threadIdx.x;
  if (idx >= B_ * H_) return;
  int b = idx / H_, j = idx % H_;
  const float* g = G + (long)b * NG_;
  float r = 1.f / (1.f + expf(-g[j]));
  float z = 1.f / (1.f + expf(-g[H_ + j]));
  float n = tanhf(g[2 * H_ + j] + r * g[3 * H_ + j]);
  half_t* hp = x_cat + (long)b * XDIM_ + XH_OFF_ + j;
  float h0 = (float)*hp;
  *hp = (half_t)((1.f - z) * n + z * h0);
}

// ---------------------------------------------------------------------------
// Per-step cross-entropy: accumulate sum of -log p(target)
// ---------------------------------------------------------------------------
__global__ __launch_bounds__(256)
void loss_kernel(const float* __restrict__ logits, const int* __restrict__ tgt,
                 float* __restrict__ acc, int t) {
  __shared__ float red[256];
  const int b = blockIdx.x, tid = threadIdx.x;
  const float* row = logits + (long)b * V_;
  float m = -3.4e38f;
  for (int v = tid; v < V_; v += 256) m = fmaxf(m, row[v]);
  red[tid] = m; __syncthreads();
  for (int o = 128; o > 0; o >>= 1) {
    if (tid < o) red[tid] = fmaxf(red[tid], red[tid + o]);
    __syncthreads();
  }
  m = red[0]; __syncthreads();
  float s = 0.f;
  for (int v = tid; v < V_; v += 256) s += expf(row[v] - m);
  red[tid] = s; __syncthreads();
  for (int o = 128; o > 0; o >>= 1) {
    if (tid < o) red[tid] += red[tid + o];
    __syncthreads();
  }
  if (tid == 0) {
    int tk = tgt[b * T_ + t];
    atomicAdd(acc, -(row[tk] - m - logf(red[0])));
  }
}

__global__ void finalize_kernel(const float* __restrict__ acc, float* __restrict__ out) {
  if (threadIdx.x == 0 && blockIdx.x == 0)
    out[0] = acc[0] / ((float)T_ * (float)B_ * (float)B_);  // sum_t mean_b / T / B
}

// ---------------------------------------------------------------------------
extern "C" void kernel_launch(void* const* d_in, const int* in_sizes, int n_in,
                              void* d_out, int out_size, void* d_ws, size_t ws_size,
                              hipStream_t stream) {
  const float* feats_f32 = (const float*)d_in[0];
  const float* sh_f32    = (const float*)d_in[1];
  const int*   tgt       = (const int*)d_in[2];
  const float* Wf   = (const float*)d_in[3];
  const float* bf   = (const float*)d_in[4];
  const float* Ws   = (const float*)d_in[5];
  const float* bs   = (const float*)d_in[6];
  const float* Wc   = (const float*)d_in[7];
  const float* bc   = (const float*)d_in[8];
  const float* vatt = (const float*)d_in[9];
  const float* emb_f32 = (const float*)d_in[10];
  const float* W_ih = (const float*)d_in[11];
  const float* W_hh = (const float*)d_in[12];
  const float* b_ih = (const float*)d_in[13];
  const float* b_hh = (const float*)d_in[14];
  const float* W_fc = (const float*)d_in[15];
  const float* b_fc = (const float*)d_in[16];
  (void)in_sizes; (void)n_in; (void)out_size; (void)ws_size;

  char* ws = (char*)d_ws;
  size_t off = 0;
  auto alloc = [&](size_t bytes) -> void* {
    void* p = (void*)(ws + off);
    off += (bytes + 255) & ~(size_t)255;
    return p;
  };
  half_t* feats_h = (half_t*)alloc((size_t)B_ * N_ * E_ * 2);      // 52 MB, L2-resident
  half_t* fproj_h = (half_t*)alloc((size_t)B_ * N_ * A_ * 2);      // 26 MB, L2-resident
  half_t* Wf_t    = (half_t*)alloc((size_t)A_ * E_ * 2);
  half_t* Ws_t    = (half_t*)alloc((size_t)A_ * SH_ * 2);
  half_t* Wc_t    = (half_t*)alloc((size_t)A_ * H_ * 2);
  half_t* Wfc_t   = (half_t*)alloc((size_t)V_ * H_ * 2);
  half_t* Wbig    = (half_t*)alloc((size_t)NG_ * XDIM_ * 2);       // 5.2 MB
  float*  bbig    = (float*)alloc((size_t)NG_ * 4);
  half_t* emb_h   = (half_t*)alloc((size_t)V_ * EMB_ * 2);
  half_t* sh_h    = (half_t*)alloc((size_t)B_ * SH_ * 2);
  float*  sproj   = (float*)alloc((size_t)B_ * A_ * 4);
  half_t* x_cat   = (half_t*)alloc((size_t)B_ * XDIM_ * 2);        // [ctx|emb|h]
  float*  G       = (float*)alloc((size_t)B_ * NG_ * 4);
  float*  lacc    = (float*)alloc(256);

  float* out = (float*)d_out;
  float* loss_out = out + (size_t)T_ * B_ * V_;

  // ---- preamble -----------------------------------------------------------
  f32_to_f16_kernel<<<4096, 256, 0, stream>>>(feats_f32, feats_h, (long)B_ * N_ * E_);
  f32_to_f16_kernel<<<512, 256, 0, stream>>>(emb_f32, emb_h, (long)V_ * EMB_);
  f32_to_f16_kernel<<<128, 256, 0, stream>>>(sh_f32, sh_h, (long)B_ * SH_);
  transpose_f32_to_f16_kernel<<<dim3(A_ / 32, E_ / 32), 256, 0, stream>>>(Wf, Wf_t, E_, A_);
  transpose_f32_to_f16_kernel<<<dim3(A_ / 32, SH_ / 32), 256, 0, stream>>>(Ws, Ws_t, SH_, A_);
  transpose_f32_to_f16_kernel<<<dim3(A_ / 32, H_ / 32), 256, 0, stream>>>(Wc, Wc_t, H_, A_);
  transpose_f32_to_f16_kernel<<<dim3((V_ + 31) / 32, H_ / 32), 256, 0, stream>>>(W_fc, Wfc_t, H_, V_);
  build_wbig_kernel<<<4096, 256, 0, stream>>>(W_ih, W_hh, b_ih, b_hh, Wbig, bbig);
  init_state_kernel<<<(B_ * H_ + 255) / 256, 256, 0, stream>>>(x_cat, lacc);

  // feat_proj = features @ Wf + bf  (one-time 6.7 GMAC GEMM, f16 out)
  wmma_gemm_kernel<<<dim3(A_ / 64, (B_ * N_) / 64), 128, 0, stream>>>(
      feats_h, Wf_t, bf, nullptr, fproj_h, B_ * N_, A_, E_, E_, E_, A_);
  // struct_proj = sh @ Ws + bs (f32, reused every step)
  wmma_gemm_kernel<<<dim3(A_ / 64, B_ / 64), 128, 0, stream>>>(
      sh_h, Ws_t, bs, sproj, nullptr, B_, A_, SH_, SH_, SH_, A_);

  // ---- sequential scan over T=256 steps -----------------------------------
  for (int t = 0; t < T_; ++t) {
    attention_kernel<<<B_, 256, 0, stream>>>(fproj_h, feats_h, sproj, bc, Wc_t,
                                             vatt, tgt, emb_h, x_cat, t);
    // G = x_cat @ Wbig^T + bbig : [128,1280] x [1280,2048]
    wmma_gemm_kernel<<<dim3(NG_ / 64, B_ / 64), 128, 0, stream>>>(
        x_cat, Wbig, bbig, G, nullptr, B_, NG_, XDIM_, XDIM_, XDIM_, NG_);
    gru_gate_kernel<<<(B_ * H_ + 255) / 256, 256, 0, stream>>>(G, x_cat);
    float* logits_t = out + (size_t)t * B_ * V_;
    // logits = h @ W_fc + b_fc : A = h slice of x_cat (lda = XDIM)
    wmma_gemm_kernel<<<dim3((V_ + 63) / 64, B_ / 64), 128, 0, stream>>>(
        x_cat + XH_OFF_, Wfc_t, b_fc, logits_t, nullptr, B_, V_, H_, XDIM_, H_, V_);
    loss_kernel<<<B_, 256, 0, stream>>>(logits_t, tgt, lacc, t);
  }
  finalize_kernel<<<1, 64, 0, stream>>>(lacc, loss_out);
}
